// SelfAttentionLayer_34282428956916
// MI455X (gfx1250) — compile-verified, matrix operand order checked by reading, not run on hardware
//
#include <hip/hip_runtime.h>

typedef unsigned short u16;
typedef unsigned int   u32;

typedef __bf16 v16bf __attribute__((ext_vector_type(16)));
typedef float  v8f   __attribute__((ext_vector_type(8)));
typedef u32    v8u   __attribute__((ext_vector_type(8)));

// ---------- helpers ----------
__device__ __forceinline__ u16 f2bf(float f) {
    u32 u = __builtin_bit_cast(u32, f);
    u += 0x7FFFu + ((u >> 16) & 1u);          // round-to-nearest-even
    return (u16)(u >> 16);
}
__device__ __forceinline__ float bf2f(u16 h) {
    u32 u = ((u32)h) << 16;
    return __builtin_bit_cast(float, u);
}
// K index (even) held by dword v of a 16-bit A/B WMMA fragment (ISA 7.12.2):
//  v in 0..3 -> k = 8*(lane/16) + 2*v ;  v in 4..7 -> k = 16 + 8*(lane/16) + 2*(v-4)
// => dwords v=0..3 are contiguous (offset lane_hi*4), v=4..7 contiguous at +8.
__device__ __forceinline__ v8u make8(uint4 a, uint4 b) {
    v8u r;
    r[0] = a.x; r[1] = a.y; r[2] = a.z; r[3] = a.w;
    r[4] = b.x; r[5] = b.y; r[6] = b.z; r[7] = b.w;
    return r;
}
// p points at the 16-dword (32 bf16) K-slab base; lh = lane>>4
__device__ __forceinline__ v8u ld_frag16(const u32* p, int lh) {
    const uint4* q = (const uint4*)p;
    return make8(q[lh], q[2 + lh]);           // two b128 loads
}
__device__ __forceinline__ v8f wmma_bf16(v8u a, v8u b, v8f c) {
    return __builtin_amdgcn_wmma_f32_16x16x32_bf16(
        false, __builtin_bit_cast(v16bf, a),
        false, __builtin_bit_cast(v16bf, b),
        (short)0, c, false, false);
}
__device__ __forceinline__ v8f zero8() {
    v8f z = {0.f, 0.f, 0.f, 0.f, 0.f, 0.f, 0.f, 0.f};
    return z;
}

// ---------- Tensor Data Mover (guarded) ----------
#if defined(__HIP_DEVICE_COMPILE__) && __has_builtin(__builtin_amdgcn_tensor_load_to_lds)
#define HAVE_TDM 1
typedef u32 v4u __attribute__((ext_vector_type(4)));
typedef int v8i __attribute__((ext_vector_type(8)));
typedef int v4i __attribute__((ext_vector_type(4)));

// Load a [rows x 16-dword] tile (row stride srcStrideDw dwords) into LDS at
// byte offset ldsOff, inserting 4 pad dwords after every 16 dwords
// (=> LDS row stride 20 dwords, matching the compute-side AST).
__device__ __forceinline__ void tdm_load_tile(const u32* gsrc, u32 ldsOff,
                                              int rows, int srcStrideDw, int dim1) {
    unsigned long long ga = (unsigned long long)(size_t)gsrc;
    v4u g0;
    g0[0] = 1u;                                   // count = 1 valid descriptor
    g0[1] = ldsOff;                               // LDS byte address
    g0[2] = (u32)ga;                              // global_addr[31:0]
    g0[3] = (u32)(ga >> 32) | (2u << 30);         // global_addr[56:32] | type=2
    v8i g1;
    // data_size=2 (4B) | pad_enable | pad_interval=3 (16 dw) | pad_amount=3 (4 dw)
    g1[0] = (int)((2u << 16) | (1u << 20) | (3u << 22) | (3u << 25));
    g1[1] = (int)(((u32)srcStrideDw & 0xFFFFu) << 16);      // tensor_dim0 lo16
    g1[2] = (int)((((u32)dim1) & 0xFFFFu) << 16);           // dim0 hi16=0 | tensor_dim1 lo16
    g1[3] = (int)(16u << 16);                               // dim1 hi16=0 | tile_dim0 = 16 dw
    g1[4] = (int)((u32)rows & 0xFFFFu);                     // tile_dim1 | tile_dim2=0
    g1[5] = (int)srcStrideDw;                               // tensor_dim0_stride lo32
    g1[6] = 0;
    g1[7] = 0;
    v4i z4 = {0, 0, 0, 0};
    v8i z8 = {0, 0, 0, 0, 0, 0, 0, 0};
    // 6-arg form (clang-23 / therock headers): g0, g1, g2, g3, g4, cpol
    __builtin_amdgcn_tensor_load_to_lds(g0, g1, z4, z4, z8, 0);
}
#endif

// ---------- f32 -> bf16 conversion (pairs) ----------
__global__ __launch_bounds__(256) void cvt_f32_bf16(const float* __restrict__ in,
                                                    u16* __restrict__ out, int npair) {
    int i = blockIdx.x * 256 + threadIdx.x;
    if (i < npair) {
        float2 f = ((const float2*)in)[i];
        u32 lo = f2bf(f.x), hi = f2bf(f.y);
        ((u32*)out)[i] = lo | (hi << 16);
    }
}

// ---------- f32 [R][C] -> bf16 transposed [C][R] (LDS-tiled) ----------
__global__ __launch_bounds__(256) void cvt_transpose(const float* __restrict__ in,
                                                     u16* __restrict__ out, int R, int C) {
    __shared__ u16 t[32][33];
    const int bx = blockIdx.x * 32;   // column tile
    const int by = blockIdx.y * 32;   // row tile
    const int tx = threadIdx.x & 31, ty = threadIdx.x >> 5;   // ty 0..7
#pragma unroll
    for (int i = 0; i < 4; ++i) {
        int r = by + ty + i * 8;
        t[ty + i * 8][tx] = f2bf(in[(size_t)r * C + bx + tx]);
    }
    __syncthreads();
#pragma unroll
    for (int i = 0; i < 4; ++i) {
        int c = bx + ty + i * 8;
        out[(size_t)c * R + by + tx] = t[tx][ty + i * 8];
    }
}

// ---------- tiled WMMA GEMM : C[M,N] = bf16(A[M,K]) * W[K,N] + bias ----------
// A staged through LDS (TDM double-buffered when available); W is passed
// PRE-TRANSPOSED (Bt[N][K]) so B fragments are contiguous b128 global loads.
// OUTMODE: 0 = f32 row-major, 1 = bf16 row-major, 2 = bf16 transposed ([N][M])
template <int OUTMODE>
__global__ __launch_bounds__(256) void gemm_bf16_kernel(
    const u32* __restrict__ A,      // [M][K/2] dwords (bf16 pairs, row-major)
    const u32* __restrict__ Bt,     // [N][K/2] dwords (weights transposed)
    const float* __restrict__ bias, // [N]
    void* __restrict__ Cout,
    int M, int N, int K) {
    constexpr int BM = 128, BN = 128, BK = 32;
    constexpr int AST = 20;                 // LDS row stride in dwords (16 + 4 pad)
    constexpr int BUFDW = BM * AST;         // dwords per A buffer
    extern __shared__ u32 lA[];             // 2 * BUFDW dwords (double buffer)

    const int tid  = threadIdx.x;
    const int lane = tid & 31;
    const int wave = tid >> 5;              // 8 waves: 4 (M) x 2 (N)
    const int wm   = wave >> 1, wn = wave & 1;
    const int lm   = lane & 15, lh = lane >> 4;
    const int m0   = blockIdx.y * BM, n0 = blockIdx.x * BN;
    const int Kd   = K >> 1;
    const int nsteps = K / BK;

    v8f acc[2][4];
#pragma unroll
    for (int mi = 0; mi < 2; ++mi)
#pragma unroll
        for (int ni = 0; ni < 4; ++ni) acc[mi][ni] = zero8();

#ifdef HAVE_TDM
    if (wave == 0)
        tdm_load_tile(A + (size_t)m0 * Kd, 0u, BM, Kd, M);
#endif

    for (int ks = 0; ks < nsteps; ++ks) {
        const int k0  = ks * BK;
        const int cur = ks & 1;
        u32* buf = lA + cur * BUFDW;
#ifdef HAVE_TDM
        if (wave == 0) __builtin_amdgcn_s_wait_tensorcnt(0);
        __syncthreads();
        if (wave == 0 && ks + 1 < nsteps)
            tdm_load_tile(A + (size_t)m0 * Kd + ((k0 + BK) >> 1),
                          (u32)((cur ^ 1) * BUFDW * 4), BM, Kd, M);
#else
        // manual staging: 512 uint4 by 256 threads (2 each), b128 in and out
#pragma unroll
        for (int i = 0; i < 2; ++i) {
            int q  = tid + i * 256;
            int r  = q >> 2, c4 = q & 3;
            uint4 d = ((const uint4*)(A + (size_t)(m0 + r) * Kd + (k0 >> 1)))[c4];
            *(uint4*)&buf[r * AST + c4 * 4] = d;
        }
        __syncthreads();
#endif
        v8u afr[2], bfr[4];
#pragma unroll
        for (int mi = 0; mi < 2; ++mi) {
            int m = wm * 32 + mi * 16 + lm;
            afr[mi] = ld_frag16(buf + m * AST, lh);          // 2x ds_load_b128
        }
#pragma unroll
        for (int ni = 0; ni < 4; ++ni) {
            int n = n0 + wn * 64 + ni * 16 + lm;
            bfr[ni] = ld_frag16(Bt + (size_t)n * Kd + (k0 >> 1), lh); // 2x b128
        }
#pragma unroll
        for (int mi = 0; mi < 2; ++mi)
#pragma unroll
            for (int ni = 0; ni < 4; ++ni)
                acc[mi][ni] = wmma_bf16(afr[mi], bfr[ni], acc[mi][ni]);
        __syncthreads();
    }

    // --- epilogue: bias add + store ---
#pragma unroll
    for (int mi = 0; mi < 2; ++mi)
#pragma unroll
        for (int ni = 0; ni < 4; ++ni) {
            int n = n0 + wn * 64 + ni * 16 + lm;
            float bv = bias[n];
#pragma unroll
            for (int r = 0; r < 8; ++r) {
                int m = m0 + wm * 32 + mi * 16 + r + (lh << 3);
                float val = acc[mi][ni][r] + bv;
                if (OUTMODE == 0)
                    ((float*)Cout)[(size_t)m * N + n] = val;
                else if (OUTMODE == 1)
                    ((u16*)Cout)[(size_t)m * N + n] = f2bf(val);
                else
                    ((u16*)Cout)[(size_t)n * M + m] = f2bf(val);
            }
        }
}

// ---------- fused attention: one wave handles one (b,h,16-query tile) ----------
// Qb/Kb : bf16 [B*S][E] (head h occupies cols h*64..h*64+63)
// Vt    : bf16 [E][B*S]  (V projection stored transposed)
// O     : bf16 [B*S][E]
#define ATT_S 2048
#define ATT_E 1024
#define SSTR  2056   // padded score row stride in bf16 (1028 dwords; stride%64=4 -> no bank conflicts)

__global__ __launch_bounds__(32) void attn_kernel(const u16* __restrict__ Qb,
                                                  const u16* __restrict__ Kb,
                                                  const u16* __restrict__ Vt,
                                                  u16* __restrict__ O) {
    extern __shared__ char smemraw[];
    u16*   sc   = (u16*)smemraw;                       // [16][SSTR] scores/probs (bf16)
    float* rinv = (float*)(smemraw + 16 * SSTR * 2);   // [16] 1/rowsum

    const int idx = blockIdx.x;
    const int qt  = idx & 127;                 // S/16 = 128 query tiles
    const int h   = (idx >> 7) & 15;
    const int b   = idx >> 11;
    const int lane = threadIdx.x;
    const int lm = lane & 15, lh = lane >> 4;

    const u32* Qu = (const u32*)Qb;            // [4096][512] dwords
    const u32* Ku = (const u32*)Kb;
    const u32* Vu = (const u32*)Vt;            // [1024][2048] dwords

    // --- Q fragments (kept in registers for the whole tile) ---
    v8u aq[2];
    const int qrow = b * ATT_S + qt * 16 + lm;
#pragma unroll
    for (int c = 0; c < 2; ++c)
        aq[c] = ld_frag16(Qu + (size_t)qrow * 512 + h * 32 + c * 16, lh);

    // --- scores: S[16 x 2048] = (Q K^T) / 8, stored bf16 in LDS ---
    for (int kt = 0; kt < ATT_S / 16; ++kt) {
        const int krow = b * ATT_S + kt * 16 + lm;
        const u32* kp = Ku + (size_t)krow * 512 + h * 32;
        v8u bk0 = ld_frag16(kp, lh);
        v8u bk1 = ld_frag16(kp + 16, lh);
        v8f s = zero8();
        s = wmma_bf16(aq[0], bk0, s);
        s = wmma_bf16(aq[1], bk1, s);
#pragma unroll
        for (int r = 0; r < 8; ++r) {
            int m = r + lh * 8;
            sc[m * SSTR + kt * 16 + lm] = f2bf(s[r] * 0.125f);   // 1/sqrt(64)
        }
    }
    __syncthreads();

    // --- softmax over each 2048-wide row (2 lanes per row, shfl_xor combine) ---
    {
        u32* scd = (u32*)sc;
        const int base = lm * (SSTR >> 1) + lh * 512;  // dword index
        float mx = -1e30f;
        for (int c = 0; c < 512; ++c) {
            u32 w = scd[base + c];
            mx = fmaxf(mx, fmaxf(bf2f((u16)w), bf2f((u16)(w >> 16))));
        }
        mx = fmaxf(mx, __shfl_xor(mx, 16, 32));
        float sum = 0.f;
        for (int c = 0; c < 512; ++c) {
            u32 w = scd[base + c];
            float e0 = __expf(bf2f((u16)w) - mx);
            float e1 = __expf(bf2f((u16)(w >> 16)) - mx);
            sum += e0 + e1;
            scd[base + c] = (u32)f2bf(e0) | ((u32)f2bf(e1) << 16);
        }
        sum += __shfl_xor(sum, 16, 32);
        if (lane < 16) rinv[lm] = 1.0f / sum;
    }
    __syncthreads();

    // --- out = P[16x2048] * V[2048x64] (V read transposed -> contiguous b128) ---
    v8f acc[4];
#pragma unroll
    for (int dt = 0; dt < 4; ++dt) acc[dt] = zero8();
    const u32* scu = (const u32*)sc;
    for (int kc = 0; kc < ATT_S / 32; ++kc) {
        v8u ap = ld_frag16(scu + lm * (SSTR >> 1) + kc * 16, lh);  // 2x ds_load_b128
#pragma unroll
        for (int dt = 0; dt < 4; ++dt) {
            const u32* vp = Vu + (size_t)(h * 64 + dt * 16 + lm) * 2048 + b * 1024 + kc * 16;
            v8u bv = ld_frag16(vp, lh);
            acc[dt] = wmma_bf16(ap, bv, acc[dt]);
        }
    }

    // --- epilogue: scale by 1/rowsum, store bf16 [B*S][E] ---
#pragma unroll
    for (int dt = 0; dt < 4; ++dt)
#pragma unroll
        for (int r = 0; r < 8; ++r) {
            int m = r + lh * 8;
            float val = acc[dt][r] * rinv[m];
            int row = b * ATT_S + qt * 16 + m;
            int col = h * 64 + dt * 16 + lm;
            O[(size_t)row * ATT_E + col] = f2bf(val);
        }
}

// ---------- host-side orchestration ----------
extern "C" void kernel_launch(void* const* d_in, const int* in_sizes, int n_in,
                              void* d_out, int out_size, void* d_ws, size_t ws_size,
                              hipStream_t stream) {
    (void)in_sizes; (void)n_in; (void)out_size; (void)ws_size;
    const float* x  = (const float*)d_in[0];
    const float* Wq = (const float*)d_in[1];
    const float* bq = (const float*)d_in[2];
    const float* Wk = (const float*)d_in[3];
    const float* bk = (const float*)d_in[4];
    const float* Wv = (const float*)d_in[5];
    const float* bv = (const float*)d_in[6];
    const float* Wf = (const float*)d_in[7];
    const float* bf = (const float*)d_in[8];

    const int M = 4096;  // B*S
    const int E = 1024;

    char* ws = (char*)d_ws;
    const size_t MiB = 1024 * 1024;
    u16* xb  = (u16*)(ws + 0);          // [4096][1024]  8 MiB
    u16* wqt = (u16*)(ws + 8  * MiB);   // transposed bf16 weights, 2 MiB each
    u16* wkt = (u16*)(ws + 10 * MiB);
    u16* wvt = (u16*)(ws + 12 * MiB);
    u16* wft = (u16*)(ws + 14 * MiB);
    u16* Qb  = (u16*)(ws + 16 * MiB);   // [4096][1024]  8 MiB
    u16* Kb  = (u16*)(ws + 24 * MiB);   // [4096][1024]
    u16* Vt  = (u16*)(ws + 32 * MiB);   // [1024][4096]  (transposed V)
    u16* AO  = (u16*)(ws + 40 * MiB);   // [4096][1024]  attention output

    // x: f32 -> bf16 ; W*: f32 [K][N] -> bf16 transposed [N][K]
    {
        int nx2 = M * E / 2;
        cvt_f32_bf16<<<(nx2 + 255) / 256, 256, 0, stream>>>(x, xb, nx2);
        dim3 tg(E / 32, E / 32), tb(256);
        cvt_transpose<<<tg, tb, 0, stream>>>(Wq, wqt, E, E);
        cvt_transpose<<<tg, tb, 0, stream>>>(Wk, wkt, E, E);
        cvt_transpose<<<tg, tb, 0, stream>>>(Wv, wvt, E, E);
        cvt_transpose<<<tg, tb, 0, stream>>>(Wf, wft, E, E);
    }

    dim3 gg(E / 128, M / 128), gb(256);
    const size_t gsm = 2 * 128 * 20 * 4;   // double-buffered A tile (padded)
    gemm_bf16_kernel<1><<<gg, gb, gsm, stream>>>((const u32*)xb, (const u32*)wqt, bq, Qb, M, E, E);
    gemm_bf16_kernel<1><<<gg, gb, gsm, stream>>>((const u32*)xb, (const u32*)wkt, bk, Kb, M, E, E);
    gemm_bf16_kernel<2><<<gg, gb, gsm, stream>>>((const u32*)xb, (const u32*)wvt, bv, Vt, M, E, E);

    // fused attention: one wave per (b,h,qtile)
    {
        const int nblocks = 2 * 16 * (2048 / 16);      // 4096
        const size_t smem = 16 * SSTR * 2 + 64;        // scores + rowinv
        attn_kernel<<<nblocks, 32, smem, stream>>>(Qb, Kb, Vt, AO);
    }

    // final projection, f32 straight into d_out
    gemm_bf16_kernel<0><<<gg, gb, gsm, stream>>>((const u32*)AO, (const u32*)wft, bf,
                                                 d_out, M, E, E);
}
